// encoderModelFlocking_76115410419809
// MI455X (gfx1250) — matrix operand order repaired and checked by reading, not compiled, not run on hardware
//
#include <hip/hip_runtime.h>
#include <hip/hip_bf16.h>

typedef __attribute__((ext_vector_type(16))) _Float16 v16h;
typedef __attribute__((ext_vector_type(8)))  float    v8f;
typedef __attribute__((ext_vector_type(4)))  float    v4f;
typedef __attribute__((ext_vector_type(4)))  uint32_t u32x4;
typedef __attribute__((ext_vector_type(8)))  uint32_t u32x8;

#define NROW 8192
#define NF   64
#define SHIFT 64.0f   // max possible dot of two tanh-bounded 64-dim vectors

union H16 { v16h h; v4f f[2]; };

// B-fragment per-lane halves: 8 at p, then 8 at p+16 (same K order as A)
__device__ __forceinline__ v16h load_contig16(const _Float16* p) {
  H16 u;
  u.f[0] = *(const v4f*)(p);
  u.f[1] = *(const v4f*)(p + 16);
  return u.h;
}
// A-fragment per-lane halves: 8 at p, then 8 at p+16
__device__ __forceinline__ v16h load_split16(const _Float16* p) {
  H16 u;
  u.f[0] = *(const v4f*)(p);
  u.f[1] = *(const v4f*)(p + 16);
  return u.h;
}

__device__ __forceinline__ v8f wmma_f16(v16h a, v16h b, v8f c) {
  return __builtin_amdgcn_wmma_f32_16x16x32_f16(false, a, false, b, (short)0, c,
                                                false, false);
}

// Low 32 bits of a generic pointer into LDS == byte offset within the WG's LDS.
__device__ __forceinline__ uint32_t lds_offset(const void* p) {
  return (uint32_t)(uintptr_t)p;
}

// Issue a 2-D TDM tile load (f16 elements) via the Tensor Data Mover.
// tile: tile_d0 elems/row x tile_d1 rows; tensor dims for OOB; row stride in elems.
__device__ __forceinline__ void tdm_load_2d(const void* gptr, uint32_t ldsoff,
                                            uint32_t tile_d0, uint32_t tile_d1,
                                            uint32_t tensor_d0, uint32_t tensor_d1,
                                            uint32_t stride0) {
  uint64_t ga = (uint64_t)(uintptr_t)gptr;
  u32x4 g0;
  g0[0] = 1u;                                    // count=1, user descriptor
  g0[1] = ldsoff;                                // lds_addr (bytes)
  g0[2] = (uint32_t)ga;                          // global_addr[31:0]
  g0[3] = ((uint32_t)(ga >> 32) & 0x01FFFFFFu)   // global_addr[56:32]
          | (2u << 30);                          // type = 2 ("image")
  u32x8 g1;
  g1[0] = (1u << 16);                            // wg_mask=0, data_size=1 (2B)
  g1[1] = (tensor_d0 & 0xFFFFu) << 16;           // tensor_dim0[15:0]
  g1[2] = (tensor_d0 >> 16) | ((tensor_d1 & 0xFFFFu) << 16);
  g1[3] = (tensor_d1 >> 16) | (tile_d0 << 16);   // tensor_dim1 hi | tile_dim0
  g1[4] = tile_d1 & 0xFFFFu;                     // tile_dim1 (tile_dim2 = 0)
  g1[5] = stride0;                               // tensor_dim0_stride[31:0]
  g1[6] = 0u;                                    // stride0 hi / dim1_stride lo
  g1[7] = 0u;                                    // dim1_stride hi
  asm volatile("tensor_load_to_lds %0, %1" :: "s"(g0), "s"(g1) : "memory");
}

// ---------------- kernel 1: per-coordinate encoder MLP -> xe (f16) + xe^T (f16)
__global__ __launch_bounds__(256) void k_encode(
    const float* __restrict__ x,
    const float* __restrict__ w1, const float* __restrict__ b1,
    const float* __restrict__ w2, const float* __restrict__ b2,
    const float* __restrict__ w3, const float* __restrict__ b3,
    _Float16* __restrict__ xe_h, _Float16* __restrict__ xeT_h) {
  int idx = blockIdx.x * blockDim.x + threadIdx.x;
  if (idx >= NROW * NF) return;
  float x0 = x[2 * idx + 0], x1 = x[2 * idx + 1];
  float h1[4], h2[4];
#pragma unroll
  for (int o = 0; o < 4; ++o)
    h1[o] = tanhf(x0 * w1[o] + x1 * w1[4 + o] + b1[o]);
#pragma unroll
  for (int o = 0; o < 4; ++o) {
    float s = b2[o];
#pragma unroll
    for (int c = 0; c < 4; ++c) s += h1[c] * w2[c * 4 + o];
    h2[o] = tanhf(s);
  }
  float s = b3[0];
#pragma unroll
  for (int c = 0; c < 4; ++c) s += h2[c] * w3[c];
  float e = tanhf(s);
  int i = idx / NF, f = idx % NF;
  xe_h[idx] = (_Float16)e;
  xeT_h[(size_t)f * NROW + i] = (_Float16)e;
}

// ---------------- kernel 2: flash pass 1 — per-row sum of exp(S - SHIFT)
// S symmetric => row sums == column-softmax denominators.
// B tiles staged once per block by the TDM (double-buffered), shared by 8 waves.
__global__ __launch_bounds__(256) void k_rowsum(const _Float16* __restrict__ xe_h,
                                                float* __restrict__ R) {
  __shared__ _Float16 sB[2][32][64];  // 8 KB, TDM double buffer
  int lane = threadIdx.x & 31;
  int w    = threadIdx.x >> 5;
  int wid  = blockIdx.x * 8 + w;
  int i0   = wid * 16;
  int m = lane & 15, hv = lane >> 4;

  v16h a0 = load_split16(xe_h + (size_t)(i0 + m) * NF + 0  + hv * 8);
  v16h a1 = load_split16(xe_h + (size_t)(i0 + m) * NF + 32 + hv * 8);

  const int NT = NROW / 32;  // 256 j-blocks
  if (w == 0) {
    tdm_load_2d(xe_h + 0 * 32 * NF, lds_offset(&sB[0][0][0]), NF, 32, NF, NROW, NF);
    tdm_load_2d(xe_h + 1 * 32 * NF, lds_offset(&sB[1][0][0]), NF, 32, NF, NROW, NF);
  }

  float acc[8];
#pragma unroll
  for (int r = 0; r < 8; ++r) acc[r] = 0.f;

  for (int k = 0; k < NT; ++k) {
    int buf = k & 1;
    if (w == 0) {
      if (k + 1 < NT) __builtin_amdgcn_s_wait_tensorcnt(1);
      else            __builtin_amdgcn_s_wait_tensorcnt(0);
    }
    __syncthreads();  // staged tile visible to all waves
#pragma unroll
    for (int jt = 0; jt < 2; ++jt) {
      v16h b0 = load_contig16(&sB[buf][jt * 16 + m][ 0 + hv * 8]);
      v16h b1 = load_contig16(&sB[buf][jt * 16 + m][32 + hv * 8]);
      v8f c = {};
      c = wmma_f16(a0, b0, c);
      c = wmma_f16(a1, b1, c);
#pragma unroll
      for (int r = 0; r < 8; ++r) acc[r] += __expf(c[r] - SHIFT);
    }
    __syncthreads();  // all waves done reading buf before it is refilled
    if (w == 0 && k + 2 < NT)
      tdm_load_2d(xe_h + (size_t)(k + 2) * 32 * NF, lds_offset(&sB[buf][0][0]),
                  NF, 32, NF, NROW, NF);
  }
  // reduce over 16 lanes within each half-wave (C layout: N = lane&15)
#pragma unroll
  for (int r = 0; r < 8; ++r) {
#pragma unroll
    for (int msk = 1; msk < 16; msk <<= 1)
      acc[r] += __shfl_xor(acc[r], msk, 32);
  }
  if (m == 0) {
#pragma unroll
    for (int r = 0; r < 8; ++r) R[i0 + 8 * hv + r] = acc[r];
  }
}

// ---------------- kernel 3: flash pass 2 — out = (exp(S-SHIFT)/R_col) @ xe
__global__ __launch_bounds__(256) void k_attn_out(
    const _Float16* __restrict__ xe_h, const _Float16* __restrict__ xeT_h,
    const float* __restrict__ R, float* __restrict__ out) {
  __shared__ _Float16 sB [2][32][64];  // xe rows  (stage-1 B), TDM staged, 8 KB
  __shared__ _Float16 sBT[2][64][32];  // xe^T cols (stage-2 B), TDM staged, 8 KB
  __shared__ _Float16 sP [8][16][32];  // per-wave probability tile transpose, 8 KB
  int lane = threadIdx.x & 31;
  int w    = threadIdx.x >> 5;
  int wid  = blockIdx.x * 8 + w;
  int i0   = wid * 16;
  int m = lane & 15, hv = lane >> 4;

  v16h a0 = load_split16(xe_h + (size_t)(i0 + m) * NF + 0  + hv * 8);
  v16h a1 = load_split16(xe_h + (size_t)(i0 + m) * NF + 32 + hv * 8);

  const int NT = NROW / 32;
  if (w == 0) {
#pragma unroll
    for (int p = 0; p < 2; ++p) {
      tdm_load_2d(xe_h + (size_t)p * 32 * NF, lds_offset(&sB[p][0][0]),
                  NF, 32, NF, NROW, NF);
      tdm_load_2d(xeT_h + (size_t)p * 32, lds_offset(&sBT[p][0][0]),
                  32, NF, NROW, NF, NROW);
    }
  }

  v8f oc[4] = {};  // D accumulators for F chunks 0..15,16..31,32..47,48..63

  for (int k = 0; k < NT; ++k) {
    int buf = k & 1;
    int j0  = k * 32;
    if (w == 0) {
      if (k + 1 < NT) __builtin_amdgcn_s_wait_tensorcnt(2);
      else            __builtin_amdgcn_s_wait_tensorcnt(0);
    }
    __syncthreads();
    // ---- stage 1: two 16x16 S tiles -> probabilities into per-wave LDS tile
#pragma unroll
    for (int jt = 0; jt < 2; ++jt) {
      v16h b0 = load_contig16(&sB[buf][jt * 16 + m][ 0 + hv * 8]);
      v16h b1 = load_contig16(&sB[buf][jt * 16 + m][32 + hv * 8]);
      v8f c = {};
      c = wmma_f16(a0, b0, c);
      c = wmma_f16(a1, b1, c);
      float rinv = 1.0f / R[j0 + jt * 16 + m];  // column normalizer
#pragma unroll
      for (int r = 0; r < 8; ++r)
        sP[w][r + 8 * hv][jt * 16 + m] = (_Float16)(__expf(c[r] - SHIFT) * rinv);
    }
    __builtin_amdgcn_wave_barrier();  // wave-private LDS; DS ops in-order per wave
    v16h pa = load_split16(&sP[w][m][hv * 8]);  // A fragment of P (16x32)
    __builtin_amdgcn_wave_barrier();
    // ---- stage 2: out_tile += P(16x32) @ xe[j0..j0+31][:]
#pragma unroll
    for (int q = 0; q < 4; ++q) {
      v16h bq = load_contig16(&sBT[buf][q * 16 + m][hv * 8]);
      oc[q] = wmma_f16(pa, bq, oc[q]);
    }
    __syncthreads();  // all waves done with buf before refill
    if (w == 0 && k + 2 < NT) {
      tdm_load_2d(xe_h + (size_t)(k + 2) * 32 * NF, lds_offset(&sB[buf][0][0]),
                  NF, 32, NF, NROW, NF);
      tdm_load_2d(xeT_h + (size_t)(k + 2) * 32, lds_offset(&sBT[buf][0][0]),
                  32, NF, NROW, NF, NROW);
    }
  }
  // D layout: element (M = r + 8*hv, N = m)
#pragma unroll
  for (int q = 0; q < 4; ++q)
#pragma unroll
    for (int r = 0; r < 8; ++r)
      out[(size_t)(i0 + 8 * hv + r) * NF + q * 16 + m] = oc[q][r];
}

// ---------------- kernel 4: reduction for mean(E) closed form
__global__ __launch_bounds__(256) void k_reduce(const float* __restrict__ out,
                                                float* __restrict__ red) {
  __shared__ float sv[NF];
  __shared__ float ss1;
  if (threadIdx.x < NF) sv[threadIdx.x] = 0.f;
  if (threadIdx.x == 0) ss1 = 0.f;
  __syncthreads();
  float loc = 0.f;
  for (size_t idx = (size_t)blockIdx.x * blockDim.x + threadIdx.x;
       idx < (size_t)NROW * NF; idx += (size_t)gridDim.x * blockDim.x) {
    float v = out[idx];
    loc += v * v;
    atomicAdd(&sv[idx & (NF - 1)], v);
  }
  atomicAdd(&ss1, loc);
  __syncthreads();
  if (threadIdx.x == 0) atomicAdd(&red[0], ss1);
  if (threadIdx.x < NF) atomicAdd(&red[1 + threadIdx.x], sv[threadIdx.x]);
}

// ---------------- kernel 5: scalar head: m -> alpha, beta
__global__ void k_scalars(const float* __restrict__ red,
                          const float* wa1, const float* ba1,
                          const float* wa2, const float* ba2,
                          const float* wb1, const float* bb1,
                          const float* wb2, const float* bb2,
                          float* __restrict__ out_ab) {
  if (threadIdx.x != 0 || blockIdx.x != 0) return;
  float s1 = red[0];
  float vv = 0.f;
  for (int f = 0; f < NF; ++f) { float v = red[1 + f]; vv += v * v; }
  const float n = (float)NROW;
  float mm = 2.f * s1 / n - 2.f * vv / (n * n);  // mean(E) closed form
  float ha0 = tanhf(mm * wa1[0] + ba1[0]);
  float ha1 = tanhf(mm * wa1[1] + ba1[1]);
  float ta  = tanhf(ha0 * wa2[0] + ha1 * wa2[1] + ba2[0]);
  float hb0 = tanhf(mm * wb1[0] + bb1[0]);
  float hb1 = tanhf(mm * wb1[1] + bb1[1]);
  float tb  = tanhf(hb0 * wb2[0] + hb1 * wb2[1] + bb2[0]);
  out_ab[0] = __expf(3.0f * ta);
  out_ab[1] = __expf(-3.0f * tb);
}

extern "C" void kernel_launch(void* const* d_in, const int* in_sizes, int n_in,
                              void* d_out, int out_size, void* d_ws, size_t ws_size,
                              hipStream_t stream) {
  (void)in_sizes; (void)n_in; (void)out_size; (void)ws_size;
  const float* x   = (const float*)d_in[0];
  const float* w1  = (const float*)d_in[1];
  const float* b1  = (const float*)d_in[2];
  const float* w2  = (const float*)d_in[3];
  const float* b2  = (const float*)d_in[4];
  const float* w3  = (const float*)d_in[5];
  const float* b3  = (const float*)d_in[6];
  const float* wa1 = (const float*)d_in[7];
  const float* ba1 = (const float*)d_in[8];
  const float* wa2 = (const float*)d_in[9];
  const float* ba2 = (const float*)d_in[10];
  const float* wb1 = (const float*)d_in[11];
  const float* bb1 = (const float*)d_in[12];
  const float* wb2 = (const float*)d_in[13];
  const float* bb2 = (const float*)d_in[14];
  float* out = (float*)d_out;

  // workspace: [R 32KB pad to 64KB][xe_h 1MB][xeT_h 1MB][red 260B]
  char* ws = (char*)d_ws;
  float*    R     = (float*)ws;
  _Float16* xe_h  = (_Float16*)(ws + (64 << 10));
  _Float16* xeT_h = (_Float16*)(ws + (64 << 10) + (size_t)NROW * NF * 2);
  float*    red   = (float*)(ws + (64 << 10) + 2 * (size_t)NROW * NF * 2);

  k_encode<<<(NROW * NF + 255) / 256, 256, 0, stream>>>(x, w1, b1, w2, b2, w3, b3,
                                                        xe_h, xeT_h);
  k_rowsum<<<NROW / 16 / 8, 256, 0, stream>>>(xe_h, R);
  k_attn_out<<<NROW / 16 / 8, 256, 0, stream>>>(xe_h, xeT_h, R, out);
  hipMemsetAsync(red, 0, (1 + NF) * sizeof(float), stream);
  k_reduce<<<512, 256, 0, stream>>>(out, red);
  k_scalars<<<1, 32, 0, stream>>>(red, wa1, ba1, wa2, ba2, wb1, bb1, wb2, bb2,
                                  out + (size_t)NROW * NF);
}
